// SelfAttentionHead_20109036879942
// MI455X (gfx1250) — matrix operand order repaired and checked by reading, compiled, not run on hardware
//
#include <hip/hip_runtime.h>

// SelfAttentionHead: B=256, T=256, C=384, H=64, softplus-masked (no softmax norm).
// Memory-bound (~120MB min traffic @ 23.3 TB/s); strategy: read x once (fused KQV),
// f16 intermediates, v_wmma_f32_16x16x32_f16 for all matmuls, V stored transposed,
// async global->LDS staging (ASYNCcnt) for the shared Q/V chunks in attention.

typedef __attribute__((ext_vector_type(16))) _Float16 v16h;
typedef __attribute__((ext_vector_type(8)))  _Float16 v8h;
typedef __attribute__((ext_vector_type(8)))  float    v8f;

#define NB 256
#define NT 256
#define NC 384
#define NH 64

__device__ __forceinline__ v16h cat8(v8h lo, v8h hi) {
    return __builtin_shufflevector(lo, hi, 0,1,2,3,4,5,6,7,8,9,10,11,12,13,14,15);
}

__device__ __forceinline__ v8f wmma_f16(v16h a, v16h b, v8f c) {
    return __builtin_amdgcn_wmma_f32_16x16x32_f16(false, a, false, b,
                                                  (short)0, c, false, false);
}

// One 16-byte async global->LDS copy (per-lane addresses), tracked by ASYNCcnt.
__device__ __forceinline__ void async_copy_b128(const void* gptr, void* lptr) {
    unsigned           lds  = (unsigned)(uintptr_t)lptr;          // addr[31:0] = LDS offset
    unsigned long long gadr = (unsigned long long)(uintptr_t)gptr;
    asm volatile("global_load_async_to_lds_b128 %0, %1, off"
                 :: "v"(lds), "v"(gadr) : "memory");
}
__device__ __forceinline__ void wait_asynccnt0() {
    asm volatile("s_wait_asynccnt 0x0" ::: "memory");
}

// ---------------------------------------------------------------------------
// Kernel 1: fused K/Q/V projection.  x:[65536,384]f32 -> k,q:[65536,64]f16,
// vT:[256][64][256]f16.  1024 blocks x 128 threads; each block = 64 rows,
// each wave = 16 rows (one WMMA M-tile), full H=64 (4 N-tiles) x 3 outputs.
// ---------------------------------------------------------------------------
__global__ __launch_bounds__(128)
void kqv_proj_kernel(const float* __restrict__ x,
                     const float* __restrict__ Wk, const float* __restrict__ bk,
                     const float* __restrict__ Wq, const float* __restrict__ bq,
                     const float* __restrict__ Wv, const float* __restrict__ bv,
                     _Float16* __restrict__ kf, _Float16* __restrict__ qf,
                     _Float16* __restrict__ vT)
{
    __shared__ __align__(16) _Float16 xs[64 * 32];      // x tile  [row][kk]
    __shared__ __align__(16) _Float16 ws[3][64 * 32];   // W tiles [n][kk] (transposed)

    const int tid     = threadIdx.x;
    const int wave    = tid >> 5;
    const int lane    = tid & 31;
    const int half    = lane >> 4;     // which 16-lane half
    const int lm      = lane & 15;
    const int rowBase = blockIdx.x * 64;

    const float* W[3] = { Wk, Wq, Wv };

    v8f acc[3][4];
    #pragma unroll
    for (int m = 0; m < 3; ++m)
        #pragma unroll
        for (int nt = 0; nt < 4; ++nt) acc[m][nt] = {};

    for (int k0 = 0; k0 < NC; k0 += 32) {
        // --- stage x tile (64x32 f32 -> f16), coalesced along kk ---
        for (int idx = tid; idx < 64 * 32; idx += 128) {
            int r  = idx >> 5;
            int kk = idx & 31;
            xs[idx] = (_Float16)x[(size_t)(rowBase + r) * NC + k0 + kk];
        }
        // --- stage W tiles transposed: ws[m][n*32+kk] = W[k0+kk][n] ---
        #pragma unroll
        for (int m = 0; m < 3; ++m)
            for (int idx = tid; idx < 64 * 32; idx += 128) {
                int kk = idx >> 6;
                int n  = idx & 63;
                ws[m][n * 32 + kk] = (_Float16)W[m][(size_t)(k0 + kk) * NH + n];
            }
        __syncthreads();

        // A fragment: lane holds row (lm), K runs [half*8,+8) and [16+half*8,+8)
        int arow = wave * 16 + lm;
        v8h alo = *(const v8h*)&xs[arow * 32 + half * 8];
        v8h ahi = *(const v8h*)&xs[arow * 32 + 16 + half * 8];
        v16h a  = cat8(alo, ahi);

        #pragma unroll
        for (int m = 0; m < 3; ++m) {
            #pragma unroll
            for (int nt = 0; nt < 4; ++nt) {
                int n = nt * 16 + lm;  // B fragment: lane = column, K runs [half*8,+8) and [16+half*8,+8)
                v16h bf = cat8(*(const v8h*)&ws[m][n * 32 + half * 8], *(const v8h*)&ws[m][n * 32 + 16 + half * 8]);
                acc[m][nt] = wmma_f16(a, bf, acc[m][nt]);
            }
        }
        __syncthreads();
    }

    // store: D layout row = i + 8*half, col = lm (+16*nt); add biases
    #pragma unroll
    for (int nt = 0; nt < 4; ++nt) {
        int h = nt * 16 + lm;
        float bK = bk[h], bQ = bq[h], bV = bv[h];
        #pragma unroll
        for (int i = 0; i < 8; ++i) {
            int rr = rowBase + wave * 16 + 8 * half + i;
            kf[(size_t)rr * NH + h] = (_Float16)(acc[0][nt][i] + bK);
            qf[(size_t)rr * NH + h] = (_Float16)(acc[1][nt][i] + bQ);
            int bb = rr >> 8, t = rr & 255;
            vT[((size_t)bb * NH + h) * NT + t] = (_Float16)(acc[2][nt][i] + bV);
        }
    }
}

// ---------------------------------------------------------------------------
// Kernel 2: per-batch attention.  S = K@Q^T * C^-0.5, causal softplus mask,
// O = S' @ V.  256 blocks x 256 threads (8 waves); wave owns 32 output rows.
// Q/V s-chunks are shared by all 8 waves -> staged once per block into LDS
// with GLOBAL_LOAD_ASYNC_TO_LDS_B128 (ASYNCcnt path), then read as fragments.
// ---------------------------------------------------------------------------
__global__ __launch_bounds__(256)
void attn_kernel(const _Float16* __restrict__ kf, const _Float16* __restrict__ qf,
                 const _Float16* __restrict__ vT, float* __restrict__ out)
{
    __shared__ __align__(16) _Float16 sTile[8][32 * 32];  // per-wave S block (16 KB)
    __shared__ __align__(16) _Float16 qs[32 * 64];        // Q chunk  [s_local][h] (4 KB)
    __shared__ __align__(16) _Float16 vs[64 * 32];        // V chunk  [h][s_local] (4 KB)

    const int tid  = threadIdx.x;
    const int wave = tid >> 5;
    const int lane = tid & 31;
    const int half = lane >> 4;
    const int lm   = lane & 15;
    const int b    = blockIdx.x;

    const _Float16* kb = kf + (size_t)b * NT * NH;
    const _Float16* qb = qf + (size_t)b * NT * NH;
    const _Float16* vb = vT + (size_t)b * NH * NT;
    const float scale  = 0.05103103588f;  // 384^-0.5

    // K A-fragments: 2 M-tiles x 2 K-steps, loaded once, reused for all s-chunks
    v16h aK[2][2];
    #pragma unroll
    for (int mt = 0; mt < 2; ++mt)
        #pragma unroll
        for (int ks = 0; ks < 2; ++ks) {
            int row = wave * 32 + mt * 16 + lm;
            v8h lo = *(const v8h*)&kb[(size_t)row * NH + ks * 32 + half * 8];
            v8h hi = *(const v8h*)&kb[(size_t)row * NH + ks * 32 + 16 + half * 8];
            aK[mt][ks] = cat8(lo, hi);
        }

    v8f o[2][4];
    #pragma unroll
    for (int mt = 0; mt < 2; ++mt)
        #pragma unroll
        for (int nt = 0; nt < 4; ++nt) o[mt][nt] = {};

    _Float16* st = &sTile[wave][0];

    for (int s0 = 0; s0 < NT; s0 += 32) {
        // ---- async-stage shared Q and V chunks into LDS (1+1 b128 per thread) ----
        __syncthreads();  // prior chunk's fragment reads complete before overwrite
        {
            // Q chunk: 32 rows x 64 h = 2048 f16; thread t copies f16[t*8 .. t*8+8)
            async_copy_b128(qb + (size_t)s0 * NH + tid * 8, &qs[tid * 8]);
            // V chunk: rows h strided by NT in vT; thread t -> h = t/4, seg = t%4
            int h = tid >> 2, seg = tid & 3;
            async_copy_b128(vb + (size_t)h * NT + s0 + seg * 8, &vs[h * 32 + seg * 8]);
            wait_asynccnt0();
        }
        __syncthreads();  // all waves' staging visible

        // ---- S block = K @ Q^T (32 rows x 32 cols) ----
        v8f s[2][2];
        #pragma unroll
        for (int mt = 0; mt < 2; ++mt)
            #pragma unroll
            for (int nt = 0; nt < 2; ++nt) s[mt][nt] = {};

        #pragma unroll
        for (int ks = 0; ks < 2; ++ks)
            #pragma unroll
            for (int nt = 0; nt < 2; ++nt) {
                int qrow = nt * 16 + lm;  // B lane = Q row (S column), local to chunk
                v16h bq_ = cat8(*(const v8h*)&qs[qrow * NH + ks * 32 + half * 8], *(const v8h*)&qs[qrow * NH + ks * 32 + 16 + half * 8]);
                #pragma unroll
                for (int mt = 0; mt < 2; ++mt)
                    s[mt][nt] = wmma_f16(aK[mt][ks], bq_, s[mt][nt]);
            }

        // ---- scale, causal mask, softplus; spill to LDS in f16 ----
        #pragma unroll
        for (int mt = 0; mt < 2; ++mt)
            #pragma unroll
            for (int nt = 0; nt < 2; ++nt)
                #pragma unroll
                for (int i = 0; i < 8; ++i) {
                    int rl = mt * 16 + 8 * half + i;
                    int cl = nt * 16 + lm;
                    int grow = wave * 32 + rl;   // query index t
                    int gcol = s0 + cl;          // key index s
                    float xv = s[mt][nt][i] * scale;
                    float sp = fmaxf(xv, 0.f) + __logf(1.f + __expf(-fabsf(xv)));
                    float val = (grow >= gcol) ? sp : 0.f;
                    st[rl * 32 + cl] = (_Float16)val;
                }

        // ---- O += S' @ V (re-stripe S via wave-private LDS) ----
        v16h vB[4];
        #pragma unroll
        for (int nt = 0; nt < 4; ++nt) {
            int h = nt * 16 + lm;  // B lane = output column h, s runs [half*8,+8) and [16+half*8,+8)
            vB[nt] = cat8(*(const v8h*)&vs[h * 32 + half * 8], *(const v8h*)&vs[h * 32 + 16 + half * 8]);
        }
        #pragma unroll
        for (int mt = 0; mt < 2; ++mt) {
            int rl = mt * 16 + lm;
            v8h lo = *(const v8h*)&st[rl * 32 + half * 8];
            v8h hi = *(const v8h*)&st[rl * 32 + 16 + half * 8];
            v16h aS = cat8(lo, hi);
            #pragma unroll
            for (int nt = 0; nt < 4; ++nt)
                o[mt][nt] = wmma_f16(aS, vB[nt], o[mt][nt]);
        }
    }

    // ---- store O (f32, coalesced along h across lanes) ----
    #pragma unroll
    for (int mt = 0; mt < 2; ++mt)
        #pragma unroll
        for (int nt = 0; nt < 4; ++nt) {
            int h = nt * 16 + lm;
            #pragma unroll
            for (int i = 0; i < 8; ++i) {
                int t = wave * 32 + mt * 16 + 8 * half + i;
                out[((size_t)b * NT + t) * NH + h] = o[mt][nt][i];
            }
        }
}

extern "C" void kernel_launch(void* const* d_in, const int* in_sizes, int n_in,
                              void* d_out, int out_size, void* d_ws, size_t ws_size,
                              hipStream_t stream) {
    (void)in_sizes; (void)n_in; (void)out_size; (void)ws_size;
    const float* x  = (const float*)d_in[0];
    const float* Wk = (const float*)d_in[1];
    const float* bk = (const float*)d_in[2];
    const float* Wq = (const float*)d_in[3];
    const float* bq = (const float*)d_in[4];
    const float* Wv = (const float*)d_in[5];
    const float* bv = (const float*)d_in[6];
    float* out = (float*)d_out;

    // workspace: k,q row-major f16 + V transposed f16  (3 x 8 MB = 24 MB)
    _Float16* kf = (_Float16*)d_ws;
    _Float16* qf = kf + (size_t)NB * NT * NH;
    _Float16* vT = qf + (size_t)NB * NT * NH;

    kqv_proj_kernel<<<dim3((NB * NT) / 64), dim3(128), 0, stream>>>(
        x, Wk, bk, Wq, bq, Wv, bv, kf, qf, vT);
    attn_kernel<<<dim3(NB), dim3(256), 0, stream>>>(kf, qf, vT, out);
}
